// MambaTrajEncoder_86251533238391
// MI455X (gfx1250) — matrix operand order repaired and checked by reading, not compiled, not run on hardware
//
#include <hip/hip_runtime.h>
#include <hip/hip_bf16.h>
#include <math.h>
#include <stdint.h>

typedef __attribute__((ext_vector_type(16))) __bf16 v16bf;
typedef __attribute__((ext_vector_type(4)))  __bf16 v4bf;
typedef __attribute__((ext_vector_type(8)))  float  v8f;
typedef __attribute__((ext_vector_type(4)))  float  v4f;

#define B_SZ     16
#define L_SEQ    2048
#define TOK      (B_SZ * L_SEQ)     // 32768 tokens
#define D_MODEL  256
#define D_INNER  512
#define D_STATE  16
#define DT_RANK  16

#define BM 128
#define BN 64
#define BK 32
#define GEMM_THREADS 256
#define APAD 8   // bf16 LDS row stride 40 elems = 80B  -> conflict-free gathers
#define FPAD 4   // f32  LDS row stride 36 elems = 144B -> conflict-free gathers

// ---------------------------------------------------------------------------
// CDNA5 async global->LDS copy (16B per lane, tracked by ASYNCcnt).
// LDS address = low 32 bits of the generic shared-space pointer (flat LDS
// aperture truncation, ISA sec 10.2).
// ---------------------------------------------------------------------------
__device__ __forceinline__ void async_ld16(void* lds_ptr, const float* g)
{
    uint32_t lds_addr = (uint32_t)(uintptr_t)lds_ptr;
    asm volatile("global_load_async_to_lds_b128 %0, %1, off"
                 :: "v"(lds_addr), "v"((unsigned long long)(uintptr_t)g)
                 : "memory");
}
__device__ __forceinline__ void wait_async()
{
    asm volatile("s_wait_asynccnt 0x0" ::: "memory");
}

// ---------------------------------------------------------------------------
// Stage A tile (BM x BK fp32) via async DMA to LDS; M,K always tile-multiples.
// ---------------------------------------------------------------------------
__device__ __forceinline__ void stage_A_async(
    const float* __restrict__ A, int lda, int m0, int k0, int tid,
    float (* __restrict__ sAf)[BK + FPAD])
{
    #pragma unroll
    for (int i = 0; i < 4; ++i) {                  // BM*BK/4 = 1024 vec4 slots
        int idx = tid + i * GEMM_THREADS;
        int r = idx >> 3, c4 = (idx & 7) << 2;     // 8 vec4 per row
        async_ld16(&sAf[r][c4], A + (size_t)(m0 + r) * lda + (k0 + c4));
    }
}

// ---------------------------------------------------------------------------
// Stage W tile (BN x BK) -> bf16 in LDS. N=48 case needs a zero-fill guard,
// so this path goes through VGPRs (b128 load, cvt_pk, b64 LDS store).
// ---------------------------------------------------------------------------
__device__ __forceinline__ void stage_B(
    const float* __restrict__ W, int ldw, int n0, int N, int k0, int tid,
    __bf16 (* __restrict__ sB)[BK + APAD])
{
    #pragma unroll
    for (int i = 0; i < 2; ++i) {                  // BN*BK/4 = 512 vec4 slots
        int idx = tid + i * GEMM_THREADS;
        int r = idx >> 3, c4 = (idx & 7) << 2;
        int gn = n0 + r;
        v4f f = {0.0f, 0.0f, 0.0f, 0.0f};
        if (gn < N)
            f = *(const v4f*)(W + (size_t)gn * ldw + (k0 + c4));
        v4bf hv;
        hv[0] = (__bf16)f[0]; hv[1] = (__bf16)f[1];
        hv[2] = (__bf16)f[2]; hv[3] = (__bf16)f[3];
        *(v4bf*)&sB[r][c4] = hv;
    }
}

// ---------------------------------------------------------------------------
// Generic GEMM:  C[M,N] = resid + alpha * (A[M,K] @ W[N,K]^T) + bias[N]
// fp32 in/out, bf16 WMMA compute (v_wmma_f32_16x16x32_bf16), f32 accumulate.
// 256 threads = 8 waves; wave (wm,wn) owns a 32x32 output region (2x2 tiles).
// Double-buffered LDS; A tiles staged by async DMA under the WMMAs.
// ---------------------------------------------------------------------------
__global__ __launch_bounds__(GEMM_THREADS)
void gemm_bf16_wmma(const float* __restrict__ A, int lda,
                    const float* __restrict__ W, int ldw,
                    float* __restrict__ C, int ldc,
                    const float* __restrict__ bias,
                    const float* __restrict__ resid,
                    const float* __restrict__ alpha_p, int alpha_idx,
                    int M, int N, int K)
{
    __shared__ __align__(16) float  sAf[2][BM][BK + FPAD];
    __shared__ __align__(16) __bf16 sB [2][BN][BK + APAD];

    const int tid  = threadIdx.x;
    const int lane = tid & 31;
    const int wave = tid >> 5;
    const int hl   = lane >> 4;     // half-wave select (wave32)
    const int lm   = lane & 15;
    const int wm   = wave & 3;      // 4 waves along M (32 rows each)
    const int wn   = wave >> 2;     // 2 waves along N (32 cols each)

    const int m0 = blockIdx.y * BM;
    const int n0 = blockIdx.x * BN;
    const int nk = K / BK;          // K always a multiple of BK here

    v8f acc[2][2];
    #pragma unroll
    for (int i = 0; i < 2; ++i)
        #pragma unroll
        for (int j = 0; j < 2; ++j)
            #pragma unroll
            for (int e = 0; e < 8; ++e)
                acc[i][j][e] = 0.0f;

    stage_A_async(A, lda, m0, 0, tid, sAf[0]);
    stage_B(W, ldw, n0, N, 0, tid, sB[0]);
    wait_async();
    __syncthreads();

    for (int kt = 0; kt < nk; ++kt) {
        const int cur = kt & 1;
        if (kt + 1 < nk) {
            stage_A_async(A, lda, m0, (kt + 1) * BK, tid, sAf[cur ^ 1]);
            stage_B(W, ldw, n0, N, (kt + 1) * BK, tid, sB[cur ^ 1]);
        }
        // Speculative prefetch of the tile after next (global_prefetch_b8)
        if (kt + 2 < nk) {
            const int kp = (kt + 2) * BK;
            if (tid < BM)
                __builtin_prefetch(A + (size_t)(m0 + tid) * lda + kp, 0, 1);
            else if (tid - BM < BN && (n0 + tid - BM) < N)
                __builtin_prefetch(W + (size_t)(n0 + tid - BM) * ldw + kp, 0, 1);
        }

        // B fragments (32x16 bf16 per tile): lane = N col, halves split K
        v16bf bfrag[2];
        #pragma unroll
        for (int nt = 0; nt < 2; ++nt) {
            int ncol = wn * 32 + nt * 16 + lm;
            #pragma unroll
            for (int e = 0; e < 16; ++e)
                bfrag[nt][e] = sB[cur][ncol][hl * 16 + e];
        }
        // A fragments (16x32): lane = M row; read fp32 from LDS, cvt to bf16.
        #pragma unroll
        for (int mt = 0; mt < 2; ++mt) {
            v16bf afrag;
            int mrow = wm * 32 + mt * 16 + lm;
            #pragma unroll
            for (int p = 0; p < 8; ++p) {
                int kb = hl * 16 + 2 * p;
                afrag[2 * p]     = (__bf16)sAf[cur][mrow][kb];
                afrag[2 * p + 1] = (__bf16)sAf[cur][mrow][kb + 1];
            }
            #pragma unroll
            for (int nt = 0; nt < 2; ++nt) {
                acc[mt][nt] = __builtin_amdgcn_wmma_f32_16x16x32_bf16(
                    false, afrag, false, bfrag[nt], (short)0, acc[mt][nt],
                    false, false);
            }
        }
        wait_async();     // A-tile DMA for kt+1 complete before hand-off
        __syncthreads();
    }

    // Epilogue: C layout per ISA (VGPR r: lanes 0-15 -> M=r, lanes 16-31 -> M=8+r)
    const float alpha = alpha_p ? alpha_p[alpha_idx] : 1.0f;
    #pragma unroll
    for (int nt = 0; nt < 2; ++nt) {
        const int col = n0 + wn * 32 + nt * 16 + lm;
        if (col >= N) continue;
        const float bv = bias ? bias[col] : 0.0f;
        #pragma unroll
        for (int mt = 0; mt < 2; ++mt) {
            const int rowbase = m0 + wm * 32 + mt * 16 + (hl ? 8 : 0);
            #pragma unroll
            for (int r = 0; r < 8; ++r) {
                const size_t o = (size_t)(rowbase + r) * ldc + col;
                float v = alpha * acc[mt][nt][r] + bv;
                if (resid) v += resid[o];
                C[o] = v;
            }
        }
    }
}

// ---------------------------------------------------------------------------
// Wave-per-token layernorm over D_MODEL=256 (8 elems/lane on wave32)
// ---------------------------------------------------------------------------
__global__ __launch_bounds__(256)
void layernorm_kernel(const float* __restrict__ x,
                      const float* __restrict__ w,
                      const float* __restrict__ b,
                      float* __restrict__ out, int ntok)
{
    const int wave = threadIdx.x >> 5;
    const int lane = threadIdx.x & 31;
    const int tok  = blockIdx.x * 8 + wave;
    if (tok >= ntok) return;
    const float* xp = x + (size_t)tok * D_MODEL;

    float v[8];
    float s = 0.0f;
    #pragma unroll
    for (int j = 0; j < 8; ++j) { v[j] = xp[lane + j * 32]; s += v[j]; }
    #pragma unroll
    for (int off = 16; off >= 1; off >>= 1) s += __shfl_xor(s, off, 32);
    const float mu = s * (1.0f / D_MODEL);

    float var = 0.0f;
    #pragma unroll
    for (int j = 0; j < 8; ++j) { float d = v[j] - mu; var += d * d; }
    #pragma unroll
    for (int off = 16; off >= 1; off >>= 1) var += __shfl_xor(var, off, 32);
    const float inv = rsqrtf(var * (1.0f / D_MODEL) + 1e-5f);

    float* op = out + (size_t)tok * D_MODEL;
    #pragma unroll
    for (int j = 0; j < 8; ++j) {
        int c = lane + j * 32;
        op[c] = (v[j] - mu) * inv * w[c] + b[c];
    }
}

// ---------------------------------------------------------------------------
// Causal depthwise conv1d (k=4) + bias + SiLU. xpart lives in xz (stride 1024).
// ---------------------------------------------------------------------------
__global__ __launch_bounds__(256)
void conv_silu_kernel(const float* __restrict__ xz,
                      const float* __restrict__ cw,
                      const float* __restrict__ cb,
                      float* __restrict__ xs)
{
    size_t idx = (size_t)blockIdx.x * blockDim.x + threadIdx.x;
    int c   = (int)(idx & (D_INNER - 1));
    int tok = (int)(idx >> 9);
    int b   = tok >> 11;        // L_SEQ = 2048
    int l   = tok & (L_SEQ - 1);

    float acc = cb[c];
    #pragma unroll
    for (int k = 0; k < 4; ++k) {
        int ls = l - 3 + k;
        if (ls >= 0)
            acc += cw[c * 4 + k] * xz[((size_t)(b * L_SEQ + ls)) * 1024 + c];
    }
    float sig = 1.0f / (1.0f + __expf(-acc));
    xs[idx] = acc * sig;
}

// ---------------------------------------------------------------------------
// dt = softplus(dbl[:, :16] @ dt_proj_w^T + dt_proj_b)   (K=16: plain VALU)
// ---------------------------------------------------------------------------
__global__ __launch_bounds__(256)
void dt_kernel(const float* __restrict__ dbl,
               const float* __restrict__ dtw,
               const float* __restrict__ dtb,
               float* __restrict__ dt)
{
    size_t idx = (size_t)blockIdx.x * blockDim.x + threadIdx.x;
    int c   = (int)(idx & (D_INNER - 1));
    int tok = (int)(idx >> 9);

    const float* dl = dbl + (size_t)tok * 48;
    float acc = dtb[c];
    #pragma unroll
    for (int r = 0; r < DT_RANK; ++r)
        acc += dl[r] * dtw[c * DT_RANK + r];
    dt[idx] = (acc > 20.0f) ? acc : log1pf(__expf(acc));
}

// ---------------------------------------------------------------------------
// Selective scan with fused Dp skip + silu(z) gating.
// One block = (batch b, 256-channel slab). Bm/Cm staged in LDS per 64-step
// chunk (reused by the block's 256 channels -> LDS broadcast instead of 256x re-read).
// y written over xpart region of xz (stride 1024) -- xpart is dead after conv.
// ---------------------------------------------------------------------------
#define TCH 64
__global__ __launch_bounds__(256)
void scan_gate_kernel(const float* __restrict__ dt,
                      const float* __restrict__ xs,
                      const float* __restrict__ dbl,
                      float* __restrict__ xz,      // z read at +512; y written at +0
                      const float* __restrict__ A_log,
                      const float* __restrict__ Dp)
{
    __shared__ float sBC[TCH][32];   // [t][0:16]=Bm, [t][16:32]=Cm

    const int tid  = threadIdx.x;
    const int b    = blockIdx.x >> 1;
    const int part = blockIdx.x & 1;
    const int c    = part * 256 + tid;

    float Av[D_STATE];
    #pragma unroll
    for (int s = 0; s < D_STATE; ++s)
        Av[s] = -__expf(A_log[c * D_STATE + s]);
    const float Dc = Dp[c];

    float hst[D_STATE];
    #pragma unroll
    for (int s = 0; s < D_STATE; ++s) hst[s] = 0.0f;

    for (int t0 = 0; t0 < L_SEQ; t0 += TCH) {
        __syncthreads();
        #pragma unroll
        for (int i = 0; i < 8; ++i) {
            int li = tid + i * 256;           // 2048 = TCH*32
            int tt = li >> 5, j = li & 31;
            sBC[tt][j] = dbl[(size_t)(b * L_SEQ + t0 + tt) * 48 + 16 + j];
        }
        __syncthreads();

        for (int ti = 0; ti < TCH; ++ti) {
            const size_t tok = (size_t)(b * L_SEQ + t0 + ti);
            const float dtv = dt[tok * D_INNER + c];
            const float xv  = xs[tok * D_INNER + c];
            const float dx  = dtv * xv;
            float yv = 0.0f;
            #pragma unroll
            for (int s = 0; s < D_STATE; ++s) {
                hst[s] = hst[s] * __expf(dtv * Av[s]) + dx * sBC[ti][s];
                yv += hst[s] * sBC[ti][16 + s];
            }
            const float zv  = xz[tok * 1024 + 512 + c];
            const float sig = 1.0f / (1.0f + __expf(-zv));
            xz[tok * 1024 + c] = (yv + xv * Dc) * (zv * sig);
        }
    }
}

// ---------------------------------------------------------------------------
extern "C" void kernel_launch(void* const* d_in, const int* in_sizes, int n_in,
                              void* d_out, int out_size, void* d_ws, size_t ws_size,
                              hipStream_t stream)
{
    (void)in_sizes; (void)n_in; (void)out_size; (void)ws_size;

    const float* seq        = (const float*)d_in[0];
    const float* inp_w      = (const float*)d_in[1];
    const float* inp_b      = (const float*)d_in[2];
    const float* ln_w       = (const float*)d_in[3];
    const float* ln_b       = (const float*)d_in[4];
    const float* in_proj_w  = (const float*)d_in[5];
    const float* conv_w     = (const float*)d_in[6];
    const float* conv_b     = (const float*)d_in[7];
    const float* x_proj_w   = (const float*)d_in[8];
    const float* dt_proj_w  = (const float*)d_in[9];
    const float* dt_proj_b  = (const float*)d_in[10];
    const float* A_log      = (const float*)d_in[11];
    const float* Dp         = (const float*)d_in[12];
    const float* out_proj_w = (const float*)d_in[13];
    const float* res_scale  = (const float*)d_in[14];
    const float* out_ln_w   = (const float*)d_in[15];
    const float* out_ln_b   = (const float*)d_in[16];

    float* ws = (float*)d_ws;
    size_t off = 0;
    float* h_buf   = ws + off; off += (size_t)TOK * D_MODEL;   // 256
    float* xn_buf  = ws + off; off += (size_t)TOK * D_MODEL;   // 256
    float* xz_buf  = ws + off; off += (size_t)TOK * 1024;      // xz (y reuses xpart half)
    float* xs_buf  = ws + off; off += (size_t)TOK * D_INNER;   // 512
    float* dbl_buf = ws + off; off += (size_t)TOK * 48;
    float* dt_buf  = ws + off; off += (size_t)TOK * D_INNER;   // 512

    const dim3 blk(GEMM_THREADS);

    // h = seq @ inp_w^T + inp_b        (M=32768, N=256, K=64)
    gemm_bf16_wmma<<<dim3(D_MODEL / BN, TOK / BM), blk, 0, stream>>>(
        seq, 64, inp_w, 64, h_buf, D_MODEL,
        inp_b, nullptr, nullptr, 0, TOK, D_MODEL, 64);

    for (int i = 0; i < 3; ++i) {
        layernorm_kernel<<<TOK / 8, 256, 0, stream>>>(
            h_buf, ln_w + i * D_MODEL, ln_b + i * D_MODEL, xn_buf, TOK);

        // xz = xn @ in_proj_w^T        (N=1024, K=256)
        gemm_bf16_wmma<<<dim3(1024 / BN, TOK / BM), blk, 0, stream>>>(
            xn_buf, D_MODEL, in_proj_w + (size_t)i * 1024 * D_MODEL, D_MODEL,
            xz_buf, 1024, nullptr, nullptr, nullptr, 0, TOK, 1024, D_MODEL);

        conv_silu_kernel<<<(TOK * D_INNER) / 256, 256, 0, stream>>>(
            xz_buf, conv_w + i * D_INNER * 4, conv_b + i * D_INNER, xs_buf);

        // dbl = xs @ x_proj_w^T        (N=48, K=512)
        gemm_bf16_wmma<<<dim3(1, TOK / BM), blk, 0, stream>>>(
            xs_buf, D_INNER, x_proj_w + (size_t)i * 48 * D_INNER, D_INNER,
            dbl_buf, 48, nullptr, nullptr, nullptr, 0, TOK, 48, D_INNER);

        dt_kernel<<<(TOK * D_INNER) / 256, 256, 0, stream>>>(
            dbl_buf, dt_proj_w + i * D_INNER * DT_RANK, dt_proj_b + i * D_INNER,
            dt_buf);

        scan_gate_kernel<<<B_SZ * 2, 256, 0, stream>>>(
            dt_buf, xs_buf, dbl_buf, xz_buf,
            A_log + (size_t)i * D_INNER * D_STATE, Dp + i * D_INNER);

        // h = h + res_scale[i] * (y @ out_proj_w^T)   (y in xz, lda=1024)
        gemm_bf16_wmma<<<dim3(D_MODEL / BN, TOK / BM), blk, 0, stream>>>(
            xz_buf, 1024, out_proj_w + (size_t)i * D_MODEL * D_INNER, D_INNER,
            h_buf, D_MODEL, nullptr, h_buf, res_scale, i, TOK, D_MODEL, D_INNER);
    }

    layernorm_kernel<<<TOK / 8, 256, 0, stream>>>(
        h_buf, out_ln_w, out_ln_b, (float*)d_out, TOK);
}
